// GTN_56040733278620
// MI455X (gfx1250) — matrix operand order repaired and checked by reading, not compiled, not run on hardware
//
#include <hip/hip_runtime.h>
#include <hip/hip_bf16.h>
#include <stdint.h>

#define N 2048
#define E 8
#define C 8

#define BM 128
#define BN 128
#define BK 32
#define PAD 8

typedef __attribute__((ext_vector_type(16))) __bf16 v16bf;
typedef __attribute__((ext_vector_type(8)))  float  v8f;

__device__ __forceinline__ unsigned short f2bf(float f) {
    union { float f; unsigned int u; } x; x.f = f;
    unsigned int u = x.u;
    unsigned int r = (u + 0x7FFFu + ((u >> 16) & 1u)) >> 16;   // RNE
    return (unsigned short)r;
}

// ---- softmax over edge dim for the 4 weight tensors: sw_all[4][C][E] ----
__global__ void softmax_kernel(const float* __restrict__ w0, const float* __restrict__ w1,
                               const float* __restrict__ w2, const float* __restrict__ w3,
                               float* __restrict__ sw_all) {
    int t = threadIdx.x;
    if (t >= 4 * C) return;
    const float* ws[4] = {w0, w1, w2, w3};
    int s = t >> 3, c = t & 7;
    const float* w = ws[s] + c * E;
    float m = w[0];
    #pragma unroll
    for (int e = 1; e < E; ++e) m = fmaxf(m, w[e]);
    float ex[E], sum = 0.f;
    #pragma unroll
    for (int e = 0; e < E; ++e) { ex[e] = __expf(w[e] - m); sum += ex[e]; }
    float inv = 1.f / sum;
    #pragma unroll
    for (int e = 0; e < E; ++e) sw_all[(s * C + c) * E + e] = ex[e] * inv;
}

// ---- gtconv mixing: 4 outputs [C,N,N] bf16 from A [N,N,E] f32 ----
__global__ void mix_kernel(const float* __restrict__ A, const float* __restrict__ sw_all,
                           unsigned short* __restrict__ o0, unsigned short* __restrict__ o1,
                           unsigned short* __restrict__ o2, unsigned short* __restrict__ o3) {
    __shared__ float sw[4 * C * E];
    int tid = threadIdx.x;
    sw[tid] = sw_all[tid];          // blockDim.x == 256 == 4*C*E
    __syncthreads();
    size_t p = (size_t)blockIdx.x * 256 + tid;   // flat (i*N + j)
    const float4* A4 = (const float4*)A;
    float4 lo = A4[p * 2 + 0];
    float4 hi = A4[p * 2 + 1];
    float av[E] = {lo.x, lo.y, lo.z, lo.w, hi.x, hi.y, hi.z, hi.w};
    unsigned short* outs[4] = {o0, o1, o2, o3};
    const size_t NN = (size_t)N * N;
    #pragma unroll
    for (int s = 0; s < 4; ++s) {
        #pragma unroll
        for (int c = 0; c < C; ++c) {
            float acc = 0.f;
            #pragma unroll
            for (int e = 0; e < E; ++e) acc = fmaf(av[e], sw[(s * C + c) * E + e], acc);
            outs[s][(size_t)c * NN + p] = f2bf(acc);
        }
    }
}

// ---- batched bf16 GEMM, f32 accumulate: Cg[c] = Ag[c] @ Bg[c], all [N,N] ----
// Register double-buffered staging + double-buffered LDS (one barrier per K-step).
__global__ __launch_bounds__(256) void gemm_bf16_kernel(
    const unsigned short* __restrict__ Ag,
    const unsigned short* __restrict__ Bg,
    float* __restrict__ Cg) {
    __shared__ unsigned short ldsA[2][BM][BK + PAD];   // row-major A tiles
    __shared__ unsigned short ldsB[2][BN][BK + PAD];   // B^T tiles: [n][k]
    const int tid  = threadIdx.x;
    const int lane = tid & 31;
    const int wave = tid >> 5;
    const int r = lane & 15;
    const int h = lane >> 4;
    const int wm = wave & 3;     // 4 waves along M -> 32-row slabs
    const int wn = wave >> 2;    // 2 waves along N -> 64-col slabs
    const int i0 = blockIdx.y * BM;
    const int j0 = blockIdx.x * BN;
    const size_t cb = (size_t)blockIdx.z * N * N;
    const unsigned short* Ac = Ag + cb;
    const unsigned short* Bc = Bg + cb;
    float* Cc = Cg + cb;

    v8f acc[2][4] = {};
    unsigned int ra[8], rb[8];   // staged tiles (2 bf16 per dword)

    auto load_regs = [&](int kk) {
        #pragma unroll
        for (int l = 0; l < 8; ++l) {
            int idx = l * 256 + tid;
            int row = idx >> 4;              // 16 dwords per 32-elem A row
            int cp  = idx & 15;
            ra[l] = *(const unsigned int*)(Ac + (size_t)(i0 + row) * N + kk + cp * 2);
        }
        #pragma unroll
        for (int l = 0; l < 8; ++l) {
            int idx = l * 256 + tid;
            int k  = idx >> 6;               // 64 dwords per 128-elem B row
            int jp = idx & 63;
            rb[l] = *(const unsigned int*)(Bc + (size_t)(kk + k) * N + j0 + jp * 2);
        }
    };
    auto store_lds = [&](int buf) {
        #pragma unroll
        for (int l = 0; l < 8; ++l) {
            int idx = l * 256 + tid;
            int row = idx >> 4;
            int cp  = idx & 15;
            *(unsigned int*)&ldsA[buf][row][cp * 2] = ra[l];
        }
        #pragma unroll
        for (int l = 0; l < 8; ++l) {
            int idx = l * 256 + tid;
            int k  = idx >> 6;
            int jp = idx & 63;
            ldsB[buf][jp * 2 + 0][k] = (unsigned short)(rb[l] & 0xFFFFu);
            ldsB[buf][jp * 2 + 1][k] = (unsigned short)(rb[l] >> 16);
        }
    };

    load_regs(0);
    store_lds(0);
    __syncthreads();

    const int NSTEP = N / BK;    // 64
    for (int s = 0; s < NSTEP; ++s) {
        const int buf = s & 1;
        if (s + 1 < NSTEP) {
            load_regs((s + 1) * BK);               // in flight across compute
            if (s + 2 < NSTEP) {                   // warm GL2/WGP$ two steps ahead
                int kk2 = (s + 2) * BK;
                __builtin_prefetch(Ac + (size_t)(i0 + (tid >> 1)) * N + kk2, 0, 1);
                __builtin_prefetch(Bc + (size_t)(kk2 + (tid >> 5)) * N + j0 + (tid & 31) * 4, 0, 1);
            }
        }

        union Frag { v16bf v; unsigned short u[16]; };
        Frag fa[2], fb[4];
        // A fragment per ISA: elems 0..7 -> K = 8h+e ; elems 8..15 -> K = 16+8h+e
        #pragma unroll
        for (int mi = 0; mi < 2; ++mi) {
            int row = wm * 32 + mi * 16 + r;
            #pragma unroll
            for (int e = 0; e < 8; ++e) {
                fa[mi].u[e]     = ldsA[buf][row][8 * h + e];
                fa[mi].u[8 + e] = ldsA[buf][row][16 + 8 * h + e];
            }
        }
        // B fragment per ISA: n = lane%16 fixed, elems 0..7 -> K = 8h+e ; elems 8..15 -> K = 16+8h+e
        #pragma unroll
        for (int ni = 0; ni < 4; ++ni) {
            int col = wn * 64 + ni * 16 + r;
            #pragma unroll
            for (int e = 0; e < 8; ++e) {
                fb[ni].u[e]     = ldsB[buf][col][8 * h + e];
                fb[ni].u[8 + e] = ldsB[buf][col][16 + 8 * h + e];
            }
        }
        #pragma unroll
        for (int mi = 0; mi < 2; ++mi)
            #pragma unroll
            for (int ni = 0; ni < 4; ++ni)
                acc[mi][ni] = __builtin_amdgcn_wmma_f32_16x16x32_bf16(
                    false, fa[mi].v, false, fb[ni].v,
                    (short)0, acc[mi][ni], false, false);

        if (s + 1 < NSTEP) store_lds(buf ^ 1);     // other buffer: no read/write conflict
        __syncthreads();
    }

    // D layout: m_local = v + 8h, n_local = lane%16
    #pragma unroll
    for (int mi = 0; mi < 2; ++mi)
        #pragma unroll
        for (int ni = 0; ni < 4; ++ni)
            #pragma unroll
            for (int v = 0; v < 8; ++v) {
                int m = i0 + wm * 32 + mi * 16 + v + 8 * h;
                int n = j0 + wn * 64 + ni * 16 + r;
                Cc[(size_t)m * N + n] = acc[mi][ni][v];
            }
}

// ---- column degree (diag excluded) -> reciprocal ----
__global__ void colsum_kernel(const float* __restrict__ H, float* __restrict__ deg_inv) {
    int j = blockIdx.x * 256 + threadIdx.x;
    int c = blockIdx.y;
    const float* Hc = H + (size_t)c * N * N;
    float s = 0.f;
    for (int i = 0; i < N; ++i) {
        float v = Hc[(size_t)i * N + j];
        s += (i == j) ? 0.f : v;
    }
    deg_inv[c * N + j] = (s == 0.f) ? 0.f : 1.f / s;
}

// ---- zero diagonal, scale column, convert to bf16 ----
__global__ void scale_kernel(const float* __restrict__ H, const float* __restrict__ deg_inv,
                             unsigned short* __restrict__ Hbf) {
    size_t idx = (size_t)blockIdx.x * 256 + threadIdx.x;
    const size_t NN = (size_t)N * N;
    int c = (int)(idx / NN);
    size_t rem = idx % NN;
    int i = (int)(rem / N), j = (int)(rem % N);
    float v = (i == j) ? 0.f : H[idx] * deg_inv[c * N + j];
    Hbf[idx] = f2bf(v);
}

// ---- mean over channels ----
__global__ void mean_kernel(const float* __restrict__ H, float* __restrict__ S) {
    size_t p = (size_t)blockIdx.x * 256 + threadIdx.x;
    const size_t NN = (size_t)N * N;
    float s = 0.f;
    #pragma unroll
    for (int c = 0; c < C; ++c) s += H[(size_t)c * NN + p];
    S[p] = s * (1.0f / C);
}

// ---- out = (S + S^T) / 2 with LDS tile transpose ----
__global__ void sym_kernel(const float* __restrict__ S, float* __restrict__ out) {
    __shared__ float t[32][33];
    int i0 = blockIdx.y * 32, j0 = blockIdx.x * 32;
    int tx = threadIdx.x, ty = threadIdx.y;   // 32 x 8
    #pragma unroll
    for (int k = 0; k < 32; k += 8) {
        int jj = ty + k;
        t[jj][tx] = S[(size_t)(j0 + jj) * N + i0 + tx];
    }
    __syncthreads();
    #pragma unroll
    for (int k = 0; k < 32; k += 8) {
        int ii = ty + k;
        float a = S[(size_t)(i0 + ii) * N + j0 + tx];
        float b = t[tx][ii];                  // S[j0+tx][i0+ii]
        out[(size_t)(i0 + ii) * N + j0 + tx] = 0.5f * (a + b);
    }
}

extern "C" void kernel_launch(void* const* d_in, const int* in_sizes, int n_in,
                              void* d_out, int out_size, void* d_ws, size_t ws_size,
                              hipStream_t stream) {
    (void)in_sizes; (void)n_in; (void)out_size; (void)ws_size;
    const float* A    = (const float*)d_in[0];
    const float* w1_0 = (const float*)d_in[1];
    const float* w2_0 = (const float*)d_in[2];
    const float* w_1  = (const float*)d_in[3];
    const float* w_2  = (const float*)d_in[4];
    float* out = (float*)d_out;

    const size_t NN = (size_t)N * N;
    char* ws = (char*)d_ws;
    size_t off = 0;
    auto alloc = [&](size_t bytes) -> char* {
        char* p = ws + off;
        off += (bytes + 255) & ~(size_t)255;
        return p;
    };
    float*          sw_all = (float*)alloc(4 * C * E * sizeof(float));
    unsigned short* a_bf   = (unsigned short*)alloc(C * NN * 2);
    unsigned short* b_bf   = (unsigned short*)alloc(C * NN * 2);
    unsigned short* w1m    = (unsigned short*)alloc(C * NN * 2);
    unsigned short* w2m    = (unsigned short*)alloc(C * NN * 2);
    float*          H32    = (float*)alloc(C * NN * 4);
    unsigned short* Hbf    = (unsigned short*)alloc(C * NN * 2);
    float*          deginv = (float*)alloc((size_t)C * N * 4);
    float*          S      = (float*)alloc(NN * 4);

    softmax_kernel<<<1, 32, 0, stream>>>(w1_0, w2_0, w_1, w_2, sw_all);
    mix_kernel<<<(unsigned)(NN / 256), 256, 0, stream>>>(A, sw_all, a_bf, b_bf, w1m, w2m);

    dim3 gg(N / BN, N / BM, C);
    gemm_bf16_kernel<<<gg, 256, 0, stream>>>(a_bf, b_bf, H32);

    unsigned short* Wm[2] = {w1m, w2m};
    for (int rr = 0; rr < 2; ++rr) {
        colsum_kernel<<<dim3(N / 256, C), 256, 0, stream>>>(H32, deginv);
        scale_kernel<<<(unsigned)((C * NN) / 256), 256, 0, stream>>>(H32, deginv, Hbf);
        gemm_bf16_kernel<<<gg, 256, 0, stream>>>(Hbf, Wm[rr], H32);
    }

    mean_kernel<<<(unsigned)(NN / 256), 256, 0, stream>>>(H32, S);
    sym_kernel<<<dim3(N / 32, N / 32), dim3(32, 8), 0, stream>>>(S, out);
}
